// SelfAttention_8203387535468
// MI455X (gfx1250) — matrix operand order repaired and checked by reading, not compiled, not run on hardware
//
#include <hip/hip_runtime.h>
#include <stdint.h>

// ---- problem constants -----------------------------------------------------
constexpr int kDim   = 1024;
constexpr int kHeads = 16;
constexpr int kHd    = 64;      // head dim
constexpr int kSeq   = 2048;
constexpr int kBatch = 2;
constexpr int kMTot  = kBatch * kSeq;   // 4096 rows (tokens)

typedef __attribute__((ext_vector_type(16))) __bf16 v16bf;
typedef __attribute__((ext_vector_type(8)))  float  v8f;

// fp32 -> bf16 round-to-nearest-even
__device__ inline unsigned short f2bf(float f) {
    unsigned int u = __float_as_uint(f);
    u += 0x7FFFu + ((u >> 16) & 1u);
    return (unsigned short)(u >> 16);
}

// ---- WMMA fragment loaders (bf16, 16x16x32) --------------------------------
// All offsets kept in 32 bits so the backend can emit saddr+voffset (GVS)
// addressing: uniform SGPR base + divergent u32 offset, no per-lane 64-bit
// address math in the hot loops.
// A fragment (16 rows x 32 K), row-major source with leading dim `ld`.
// Lane l<16: row=l, K=0..15 contiguous ; lane l>=16: row=l-16, K=16..31.
__device__ inline v16bf load_a(const unsigned short* p, int ld, int r0, int k0, int lane) {
    int r = lane & 15, h = lane >> 4;
    unsigned off = (unsigned)((r0 + r) * ld + k0 + h * 16);
    union { uint4 q[2]; v16bf v; } u;
    u.q[0] = *reinterpret_cast<const uint4*>(p + off);
    u.q[1] = *reinterpret_cast<const uint4*>(p + off + 8);
    return u.v;
}

// B fragment (32 K x 16 cols) loaded from B^T stored row-major (col-major B):
// lane l<16: col=l, K=0..15 contiguous ; lane l>=16: col=l-16, K=16..31.
__device__ inline v16bf load_b(const unsigned short* p, int ld, int n0, int k0, int lane) {
    int n = lane & 15, h = lane >> 4;
    unsigned off = (unsigned)((n0 + n) * ld + k0 + h * 16);
    union { uint4 q[2]; v16bf v; } u;
    u.q[0] = *reinterpret_cast<const uint4*>(p + off);
    u.q[1] = *reinterpret_cast<const uint4*>(p + off + 8);
    return u.v;
}

__device__ inline v8f wmma_bf16(v16bf a, v16bf b, v8f c) {
    return __builtin_amdgcn_wmma_f32_16x16x32_bf16(
        /*neg_a=*/false, a, /*neg_b=*/false, b,
        /*c_mod=*/(short)0, c, /*reuse_a=*/false, /*reuse_b=*/false);
}

// ---- kernel 1: fp32 -> bf16 conversion (x4 vectorized) ---------------------
__global__ void cvt_bf16(const float* __restrict__ src, unsigned short* __restrict__ dst, int n4) {
    int i = blockIdx.x * blockDim.x + threadIdx.x;
    if (i < n4) {
        float4 f = reinterpret_cast<const float4*>(src)[i];
        union { unsigned short s[4]; uint2 u; } o;
        o.s[0] = f2bf(f.x); o.s[1] = f2bf(f.y); o.s[2] = f2bf(f.z); o.s[3] = f2bf(f.w);
        reinterpret_cast<uint2*>(dst)[i] = o.u;
    }
}

// ---- kernel 2: QKV projection + head scatter -------------------------------
// qkv[m,n] = sum_k x[m,k] * w_qkv[n,k].  Wave computes a 32x64 tile.
// Ping-pong fragment buffers over K: loads for chunk k+64 issue while chunk
// k+32 computes.  Scatter: n<1024 -> q (pre-scaled by D^-0.5 * log2 e),
// n<2048 -> k, else v^T.
__global__ void __launch_bounds__(256) qkv_gemm(
        const unsigned short* __restrict__ xb, const unsigned short* __restrict__ wb,
        unsigned short* __restrict__ qO, unsigned short* __restrict__ kO,
        unsigned short* __restrict__ vtO) {
    const int lane = threadIdx.x & 31;
    const int wid  = blockIdx.x * (blockDim.x >> 5) + (threadIdx.x >> 5);
    const int NB   = (3 * kDim) / 64;           // 48 column tiles of 64
    const int m0   = (wid / NB) * 32;
    const int n0   = (wid % NB) * 64;
    v8f acc[2][4] = {};

    v16bf A0[2], A1[2], B0[2], B1[2], B2[2], B3[2];
    A0[0] = load_a(xb, kDim, m0,      0, lane);
    A1[0] = load_a(xb, kDim, m0 + 16, 0, lane);
    B0[0] = load_b(wb, kDim, n0,      0, lane);
    B1[0] = load_b(wb, kDim, n0 + 16, 0, lane);
    B2[0] = load_b(wb, kDim, n0 + 32, 0, lane);
    B3[0] = load_b(wb, kDim, n0 + 48, 0, lane);

#pragma unroll 1
    for (int k0 = 0; k0 < kDim; k0 += 64) {
        // load buffer 1 <- k0+32
        A0[1] = load_a(xb, kDim, m0,      k0 + 32, lane);
        A1[1] = load_a(xb, kDim, m0 + 16, k0 + 32, lane);
        B0[1] = load_b(wb, kDim, n0,      k0 + 32, lane);
        B1[1] = load_b(wb, kDim, n0 + 16, k0 + 32, lane);
        B2[1] = load_b(wb, kDim, n0 + 32, k0 + 32, lane);
        B3[1] = load_b(wb, kDim, n0 + 48, k0 + 32, lane);
        // compute buffer 0
        acc[0][0] = wmma_bf16(A0[0], B0[0], acc[0][0]);
        acc[0][1] = wmma_bf16(A0[0], B1[0], acc[0][1]);
        acc[0][2] = wmma_bf16(A0[0], B2[0], acc[0][2]);
        acc[0][3] = wmma_bf16(A0[0], B3[0], acc[0][3]);
        acc[1][0] = wmma_bf16(A1[0], B0[0], acc[1][0]);
        acc[1][1] = wmma_bf16(A1[0], B1[0], acc[1][1]);
        acc[1][2] = wmma_bf16(A1[0], B2[0], acc[1][2]);
        acc[1][3] = wmma_bf16(A1[0], B3[0], acc[1][3]);
        // load buffer 0 <- k0+64 (wraps to 0 on last iter; harmless reload)
        const int kn = (k0 + 64) & (kDim - 1);
        A0[0] = load_a(xb, kDim, m0,      kn, lane);
        A1[0] = load_a(xb, kDim, m0 + 16, kn, lane);
        B0[0] = load_b(wb, kDim, n0,      kn, lane);
        B1[0] = load_b(wb, kDim, n0 + 16, kn, lane);
        B2[0] = load_b(wb, kDim, n0 + 32, kn, lane);
        B3[0] = load_b(wb, kDim, n0 + 48, kn, lane);
        // compute buffer 1
        acc[0][0] = wmma_bf16(A0[1], B0[1], acc[0][0]);
        acc[0][1] = wmma_bf16(A0[1], B1[1], acc[0][1]);
        acc[0][2] = wmma_bf16(A0[1], B2[1], acc[0][2]);
        acc[0][3] = wmma_bf16(A0[1], B3[1], acc[0][3]);
        acc[1][0] = wmma_bf16(A1[1], B0[1], acc[1][0]);
        acc[1][1] = wmma_bf16(A1[1], B1[1], acc[1][1]);
        acc[1][2] = wmma_bf16(A1[1], B2[1], acc[1][2]);
        acc[1][3] = wmma_bf16(A1[1], B3[1], acc[1][3]);
    }

    const float qscale = (1.0f / 32.0f) * 1.44269504088896340736f; // D^-0.5 * log2(e)
#pragma unroll
    for (int r = 0; r < 2; ++r) {
#pragma unroll
        for (int t = 0; t < 4; ++t) {
            int n = n0 + t * 16 + (lane & 15);
#pragma unroll
            for (int j = 0; j < 8; ++j) {
                int m  = m0 + r * 16 + ((lane >> 4) << 3) + j;
                int bi = m >> 11, l = m & (kSeq - 1);
                float val = acc[r][t][j];
                if (n < kDim) {
                    int h = n >> 6, d = n & 63;
                    unsigned off = (unsigned)((((bi * kHeads + h) * kSeq + l) << 6) + d);
                    qO[off] = f2bf(val * qscale);
                } else if (n < 2 * kDim) {
                    int nn = n - kDim, h = nn >> 6, d = nn & 63;
                    unsigned off = (unsigned)((((bi * kHeads + h) * kSeq + l) << 6) + d);
                    kO[off] = f2bf(val);
                } else {
                    int nn = n - 2 * kDim, h = nn >> 6, d = nn & 63;
                    unsigned off = (unsigned)(((bi * kHeads + h) * kHd + d) * kSeq + l);
                    vtO[off] = f2bf(val);
                }
            }
        }
    }
}

// ---- kernel 3: flash attention (online softmax, base-2 domain) -------------
// block = 4 waves x 32 lanes; each wave owns 16 query rows; 64-key KV tiles.
__global__ void __launch_bounds__(128) flash_attn(
        const unsigned short* __restrict__ q, const unsigned short* __restrict__ k,
        const unsigned short* __restrict__ vt, unsigned short* __restrict__ o) {
    __shared__ unsigned short ldsP[4][16][64];  // wave-private P staging tiles
    const int lane = threadIdx.x & 31;
    const int wave = threadIdx.x >> 5;
    const int bh = blockIdx.x >> 5;             // 32 q-tiles of 64 per (b,h)
    const int qt = blockIdx.x & 31;
    const int bi = bh >> 4, h = bh & 15;
    const int q0 = qt * 64 + wave * 16;
    const unsigned short* qp = q  + (size_t)bh * kSeq * kHd;
    const unsigned short* kp = k  + (size_t)bh * kSeq * kHd;
    const unsigned short* vp = vt + (size_t)bh * kHd * kSeq;

    // Q rows live in registers for the whole kernel (pre-scaled by scale*log2e)
    v16bf qa0 = load_a(qp, kHd, q0, 0,  lane);
    v16bf qa1 = load_a(qp, kHd, q0, 32, lane);

    float m_i[8], l_i[8];
    v8f oAcc[4] = {};
#pragma unroll
    for (int j = 0; j < 8; ++j) { m_i[j] = -1e30f; l_i[j] = 0.0f; }

    unsigned short* myP = &ldsP[wave][0][0];

#pragma unroll 1
    for (int kv = 0; kv < kSeq; kv += 64) {
        if (kv + 64 < kSeq)   // hint next K tile toward cache (global_prefetch)
            __builtin_prefetch(kp + (unsigned)((kv + 64 + (lane & 15)) * kHd), 0, 0);

        // ---- S = Q·K^T over 4 key sub-tiles (contraction hd=64, 8 WMMAs)
        v8f sf[4];
#pragma unroll
        for (int s = 0; s < 4; ++s) {
            v8f c = {};
            v16bf kb = load_b(kp, kHd, kv + s * 16, 0, lane);
            c = wmma_bf16(qa0, kb, c);
            kb = load_b(kp, kHd, kv + s * 16, 32, lane);
            sf[s] = wmma_bf16(qa1, kb, c);
        }

        // ---- issue all V fragment loads now; latency hides behind softmax
        v16bf vb[4][2];
#pragma unroll
        for (int t = 0; t < 4; ++t)
#pragma unroll
            for (int c = 0; c < 2; ++c)
                vb[t][c] = load_b(vp, kSeq, t * 16, kv + c * 32, lane);

        // ---- online softmax over the 64-key tile
        float alpha[8];
#pragma unroll
        for (int j = 0; j < 8; ++j) {
            float rmax = fmaxf(fmaxf(sf[0][j], sf[1][j]), fmaxf(sf[2][j], sf[3][j]));
            rmax = fmaxf(rmax, __shfl_xor(rmax, 1, 32));
            rmax = fmaxf(rmax, __shfl_xor(rmax, 2, 32));
            rmax = fmaxf(rmax, __shfl_xor(rmax, 4, 32));
            rmax = fmaxf(rmax, __shfl_xor(rmax, 8, 32));
            float mnew = fmaxf(m_i[j], rmax);
            alpha[j] = exp2f(m_i[j] - mnew);            // v_exp_f32 (native 2^x)
            float p0 = exp2f(sf[0][j] - mnew);
            float p1 = exp2f(sf[1][j] - mnew);
            float p2 = exp2f(sf[2][j] - mnew);
            float p3 = exp2f(sf[3][j] - mnew);
            float rs = (p0 + p1) + (p2 + p3);
            rs += __shfl_xor(rs, 1, 32);
            rs += __shfl_xor(rs, 2, 32);
            rs += __shfl_xor(rs, 4, 32);
            rs += __shfl_xor(rs, 8, 32);
            l_i[j] = l_i[j] * alpha[j] + rs;
            m_i[j] = mnew;
            // C-layout -> row-major LDS (lane holds column n, 8 rows)
            int row = ((lane >> 4) << 3) + j;
            int col = lane & 15;
            myP[row * 64 + col]      = f2bf(p0);
            myP[row * 64 + col + 16] = f2bf(p1);
            myP[row * 64 + col + 32] = f2bf(p2);
            myP[row * 64 + col + 48] = f2bf(p3);
        }
        // rescale running O accumulator per row
#pragma unroll
        for (int t = 0; t < 4; ++t)
#pragma unroll
            for (int j = 0; j < 8; ++j) oAcc[t][j] *= alpha[j];

        // reload P as A fragments (ds_load_b128), then P·V (8 WMMAs)
        v16bf pa0 = load_a(myP, 64, 0, 0,  lane);
        v16bf pa1 = load_a(myP, 64, 0, 32, lane);
#pragma unroll
        for (int t = 0; t < 4; ++t) {
            oAcc[t] = wmma_bf16(pa0, vb[t][0], oAcc[t]);
            oAcc[t] = wmma_bf16(pa1, vb[t][1], oAcc[t]);
        }
    }

    // normalize and write O back to [B, L, D] bf16 (input to out-projection)
#pragma unroll
    for (int t = 0; t < 4; ++t) {
        int d = t * 16 + (lane & 15);
#pragma unroll
        for (int j = 0; j < 8; ++j) {
            int m = q0 + ((lane >> 4) << 3) + j;
            float val = oAcc[t][j] / l_i[j];
            unsigned off = (unsigned)((bi * kSeq + m) * kDim + h * kHd + d);
            o[off] = f2bf(val);
        }
    }
}

// ---- kernel 4: output projection + bias (32x64 tile, ping-pong) ------------
__global__ void __launch_bounds__(256) out_gemm(
        const unsigned short* __restrict__ ob, const unsigned short* __restrict__ wb,
        const float* __restrict__ bias, float* __restrict__ out) {
    const int lane = threadIdx.x & 31;
    const int wid  = blockIdx.x * (blockDim.x >> 5) + (threadIdx.x >> 5);
    const int NB   = kDim / 64;                 // 16
    const int m0   = (wid / NB) * 32;
    const int n0   = (wid % NB) * 64;
    v8f acc[2][4] = {};

    v16bf A0[2], A1[2], B0[2], B1[2], B2[2], B3[2];
    A0[0] = load_a(ob, kDim, m0,      0, lane);
    A1[0] = load_a(ob, kDim, m0 + 16, 0, lane);
    B0[0] = load_b(wb, kDim, n0,      0, lane);
    B1[0] = load_b(wb, kDim, n0 + 16, 0, lane);
    B2[0] = load_b(wb, kDim, n0 + 32, 0, lane);
    B3[0] = load_b(wb, kDim, n0 + 48, 0, lane);

#pragma unroll 1
    for (int k0 = 0; k0 < kDim; k0 += 64) {
        A0[1] = load_a(ob, kDim, m0,      k0 + 32, lane);
        A1[1] = load_a(ob, kDim, m0 + 16, k0 + 32, lane);
        B0[1] = load_b(wb, kDim, n0,      k0 + 32, lane);
        B1[1] = load_b(wb, kDim, n0 + 16, k0 + 32, lane);
        B2[1] = load_b(wb, kDim, n0 + 32, k0 + 32, lane);
        B3[1] = load_b(wb, kDim, n0 + 48, k0 + 32, lane);

        acc[0][0] = wmma_bf16(A0[0], B0[0], acc[0][0]);
        acc[0][1] = wmma_bf16(A0[0], B1[0], acc[0][1]);
        acc[0][2] = wmma_bf16(A0[0], B2[0], acc[0][2]);
        acc[0][3] = wmma_bf16(A0[0], B3[0], acc[0][3]);
        acc[1][0] = wmma_bf16(A1[0], B0[0], acc[1][0]);
        acc[1][1] = wmma_bf16(A1[0], B1[0], acc[1][1]);
        acc[1][2] = wmma_bf16(A1[0], B2[0], acc[1][2]);
        acc[1][3] = wmma_bf16(A1[0], B3[0], acc[1][3]);

        const int kn = (k0 + 64) & (kDim - 1);
        A0[0] = load_a(ob, kDim, m0,      kn, lane);
        A1[0] = load_a(ob, kDim, m0 + 16, kn, lane);
        B0[0] = load_b(wb, kDim, n0,      kn, lane);
        B1[0] = load_b(wb, kDim, n0 + 16, kn, lane);
        B2[0] = load_b(wb, kDim, n0 + 32, kn, lane);
        B3[0] = load_b(wb, kDim, n0 + 48, kn, lane);

        acc[0][0] = wmma_bf16(A0[1], B0[1], acc[0][0]);
        acc[0][1] = wmma_bf16(A0[1], B1[1], acc[0][1]);
        acc[0][2] = wmma_bf16(A0[1], B2[1], acc[0][2]);
        acc[0][3] = wmma_bf16(A0[1], B3[1], acc[0][3]);
        acc[1][0] = wmma_bf16(A1[1], B0[1], acc[1][0]);
        acc[1][1] = wmma_bf16(A1[1], B1[1], acc[1][1]);
        acc[1][2] = wmma_bf16(A1[1], B2[1], acc[1][2]);
        acc[1][3] = wmma_bf16(A1[1], B3[1], acc[1][3]);
    }

#pragma unroll
    for (int r = 0; r < 2; ++r) {
#pragma unroll
        for (int t = 0; t < 4; ++t) {
            int n = n0 + t * 16 + (lane & 15);
            float bn = bias[n];
#pragma unroll
            for (int j = 0; j < 8; ++j) {
                int m = m0 + r * 16 + ((lane >> 4) << 3) + j;
                out[(unsigned)(m * kDim + n)] = acc[r][t][j] + bn;
            }
        }
    }
}

// ---- launcher --------------------------------------------------------------
extern "C" void kernel_launch(void* const* d_in, const int* in_sizes, int n_in,
                              void* d_out, int out_size, void* d_ws, size_t ws_size,
                              hipStream_t stream) {
    const float* x      = (const float*)d_in[0];
    const float* w_qkv  = (const float*)d_in[1];
    const float* w_out  = (const float*)d_in[2];
    const float* b_out  = (const float*)d_in[3];
    float* out = (float*)d_out;

    // workspace carve-up (bf16 intermediates), ~48 MB total
    char* ws = (char*)d_ws;
    size_t off = 0;
    unsigned short* xb  = (unsigned short*)(ws + off); off += (size_t)kMTot * kDim * 2;          // 8 MB
    unsigned short* wqb = (unsigned short*)(ws + off); off += (size_t)3 * kDim * kDim * 2;       // 6 MB
    unsigned short* wob = (unsigned short*)(ws + off); off += (size_t)kDim * kDim * 2;           // 2 MB
    unsigned short* qW  = (unsigned short*)(ws + off); off += (size_t)kMTot * kDim * 2;          // 8 MB
    unsigned short* kW  = (unsigned short*)(ws + off); off += (size_t)kMTot * kDim * 2;          // 8 MB
    unsigned short* vtW = (unsigned short*)(ws + off); off += (size_t)kMTot * kDim * 2;          // 8 MB
    unsigned short* oW  = (unsigned short*)(ws + off); off += (size_t)kMTot * kDim * 2;          // 8 MB
    (void)ws_size; (void)in_sizes; (void)n_in; (void)out_size;

    const int nx4 = kMTot * kDim / 4, nwq4 = 3 * kDim * kDim / 4, nwo4 = kDim * kDim / 4;
    cvt_bf16<<<(nx4  + 255) / 256, 256, 0, stream>>>(x,     xb,  nx4);
    cvt_bf16<<<(nwq4 + 255) / 256, 256, 0, stream>>>(w_qkv, wqb, nwq4);
    cvt_bf16<<<(nwo4 + 255) / 256, 256, 0, stream>>>(w_out, wob, nwo4);

    // QKV: (4096/32) x 48 = 6144 waves / 8 per block
    qkv_gemm<<<768, 256, 0, stream>>>(xb, wqb, qW, kW, vtW);

    // Flash attention: B*H*(L/64) = 1024 blocks of 4 waves
    flash_attn<<<kBatch * kHeads * (kSeq / 64), 128, 0, stream>>>(qW, kW, vtW, oW);

    // Out projection: (4096/32) x 16 = 2048 waves / 8 per block
    out_gemm<<<256, 256, 0, stream>>>(oW, wob, b_out, out);
}
